// SpecificAtomKernelLayer_18863496364430
// MI455X (gfx1250) — matrix operand order repaired and checked by reading, not compiled, not run on hardware
//
#include <hip/hip_runtime.h>

// out[n] = sum_m exp(-||Xnew_n - Xtr_m||^2 / 2) * w[m]
//        = exp(-x2_n/2) * sum_m exp(cross_nm) * [ w_m * exp(-y2_m/2) ]
//
// v3: factor the Gaussian so the WMMA inner loop's epilogue is just
// exp(cross) * wy (3 VALU + 1 trans per element). All f32->f16 hi/lo
// splitting and norm/weight folding happens once in a prep pass.

typedef __attribute__((ext_vector_type(16))) _Float16 v16h;
typedef __attribute__((ext_vector_type(8)))  _Float16 v8h;
typedef __attribute__((ext_vector_type(4)))  _Float16 v4h;
typedef __attribute__((ext_vector_type(8)))  float    v8f;

#define KDIM 128

__device__ __forceinline__ v8h ld8h(const _Float16* p) {
    return *reinterpret_cast<const v8h*>(p);
}
__device__ __forceinline__ v16h cat16(v8h a, v8h b) {
    return __builtin_shufflevector(a, b, 0, 1, 2, 3, 4, 5, 6, 7,
                                         8, 9, 10, 11, 12, 13, 14, 15);
}

// ---------------------------------------------------------------------------
// Prep: one wave per row. Split f32 row into f16 hi + f16 lo residual arrays;
// fold the row's Gaussian norm factor (and optional weight) into one scalar:
//   scale[row] = (wvec ? wvec[row] : 1) * exp(-0.5 * ||row||^2)
// ---------------------------------------------------------------------------
__global__ __launch_bounds__(256)
void prep_split_kernel(const float* __restrict__ X,
                       const float* __restrict__ wvec,
                       _Float16* __restrict__ hi,
                       _Float16* __restrict__ lo,
                       float* __restrict__ scale,
                       int nrows)
{
    const int gid  = blockIdx.x * blockDim.x + threadIdx.x;
    const int row  = gid >> 5;
    const int lane = gid & 31;
    if (row >= nrows) return;

    const size_t base = (size_t)row * KDIM + lane * 4;
    const float4 v = *reinterpret_cast<const float4*>(X + base);

    float sq = v.x * v.x;
    sq = fmaf(v.y, v.y, sq);
    sq = fmaf(v.z, v.z, sq);
    sq = fmaf(v.w, v.w, sq);

    _Float16 hx = (_Float16)v.x, hy = (_Float16)v.y;
    _Float16 hz = (_Float16)v.z, hw = (_Float16)v.w;
    v4h hv = {hx, hy, hz, hw};
    v4h lv = {(_Float16)(v.x - (float)hx), (_Float16)(v.y - (float)hy),
              (_Float16)(v.z - (float)hz), (_Float16)(v.w - (float)hw)};
    *reinterpret_cast<v4h*>(hi + base) = hv;
    *reinterpret_cast<v4h*>(lo + base) = lv;

    #pragma unroll
    for (int o = 16; o > 0; o >>= 1) sq += __shfl_xor(sq, o, 32);
    if (lane == 0) {
        const float wv = wvec ? wvec[row] : 1.0f;
        scale[row] = wv * __expf(-0.5f * sq);
    }
}

// ---------------------------------------------------------------------------
// Main: 512 blocks x 4 waves. Block owns 16 Xnew rows (A in registers);
// waves stripe the train rows in 16-row B tiles. Per tile: 12 split-f16
// WMMAs (two independent accumulator chains) + exp(cross)*wy accumulate.
// Row factor exp(-x2/2) applied once at the end. Fixed-order LDS reduction.
// ---------------------------------------------------------------------------
__global__ __launch_bounds__(128)
void rbf_main_kernel(const _Float16* __restrict__ Ah, const _Float16* __restrict__ Al,
                     const _Float16* __restrict__ Bh, const _Float16* __restrict__ Bl,
                     const float* __restrict__ exn,   // exp(-x2/2) per Xnew row
                     const float* __restrict__ wy,    // w * exp(-y2/2) per Xtr row
                     float* __restrict__ out,
                     int Ntrain)
{
    const int lane = threadIdx.x & 31;
    const int wid  = threadIdx.x >> 5;
    const int h    = lane >> 4;          // lane half (ISA K-split)
    const int l15  = lane & 15;
    const int row0 = blockIdx.x * 16;

    // A fragments: contiguous-k 16x32 layout (same k order as B). Lane (M=l15, half h), chunk c:
    // 16 contiguous halves from 32c+16h -> two 16B loads, shuffled.
    v16h ah[4], al[4];
    {
        const _Float16* arh = Ah + (size_t)(row0 + l15) * KDIM;
        const _Float16* arl = Al + (size_t)(row0 + l15) * KDIM;
        #pragma unroll
        for (int c = 0; c < 4; ++c) {
            const int o1 = 32 * c + 16 * h;
            ah[c] = cat16(ld8h(arh + o1), ld8h(arh + o1 + 8));
            al[c] = cat16(ld8h(arl + o1), ld8h(arl + o1 + 8));
        }
    }

    float p[8] = {0.f, 0.f, 0.f, 0.f, 0.f, 0.f, 0.f, 0.f};

    const int iters = Ntrain >> 6;  // 16 m per tile x 4 waves
    for (int it = 0; it < iters; ++it) {
        const int m0 = ((it << 2) + wid) << 4;

        // B fragments: contiguous-k 32x16 layout (same k order as A). Lane (N=l15, half h), chunk c:
        // 16 contiguous halves from 32c+16h -> two 16B loads, shuffled.
        const _Float16* brh = Bh + (size_t)(m0 + l15) * KDIM + 16 * h;
        const _Float16* brl = Bl + (size_t)(m0 + l15) * KDIM + 16 * h;
        v16h bh[4], bl[4];
        #pragma unroll
        for (int c = 0; c < 4; ++c) {
            bh[c] = cat16(ld8h(brh + 32 * c), ld8h(brh + 32 * c + 8));
            bl[c] = cat16(ld8h(brl + 32 * c), ld8h(brl + 32 * c + 8));
        }
        const float wv = wy[m0 + l15];  // folded weight * exp(-y2/2)

        if (it + 1 < iters)
            __builtin_prefetch(brh + 64 * KDIM, 0, 1);  // next tile for this wave

        // Two independent accumulator chains: hi*hi and (hi*lo + lo*hi).
        v8f acc0 = {};
        v8f acc1 = {};
        #pragma unroll
        for (int c = 0; c < 4; ++c) {
            acc0 = __builtin_amdgcn_wmma_f32_16x16x32_f16(
                false, ah[c], false, bh[c], (short)0, acc0, false, false);
            acc1 = __builtin_amdgcn_wmma_f32_16x16x32_f16(
                false, ah[c], false, bl[c], (short)0, acc1, false, false);
            acc1 = __builtin_amdgcn_wmma_f32_16x16x32_f16(
                false, al[c], false, bh[c], (short)0, acc1, false, false);
        }

        // Minimal epilogue: p += exp(cross) * wy   (3 VALU + 1 trans / element)
        #pragma unroll
        for (int r = 0; r < 8; ++r) {
            p[r] = fmaf(__expf(acc0[r] + acc1[r]), wv, p[r]);  // v_exp_f32
        }
    }

    // Deterministic reduction: 16 columns per lane-half, then 4 waves.
    __shared__ float red[4][32][8];
    #pragma unroll
    for (int r = 0; r < 8; ++r) red[wid][lane][r] = p[r];
    __syncthreads();

    if (threadIdx.x < 16) {
        const int t  = threadIdx.x;
        const int th = t >> 3;
        const int tr = t & 7;
        float s = 0.0f;
        for (int ww = 0; ww < 4; ++ww)
            for (int l = 0; l < 16; ++l)
                s += red[ww][16 * th + l][tr];
        out[row0 + t] = exn[row0 + t] * s;   // apply exp(-x2/2) once per row
    }
}

extern "C" void kernel_launch(void* const* d_in, const int* in_sizes, int n_in,
                              void* d_out, int out_size, void* d_ws, size_t ws_size,
                              hipStream_t stream) {
    (void)n_in; (void)out_size; (void)ws_size;
    const float* Xnew = (const float*)d_in[0];
    const float* Xtr  = (const float*)d_in[1];
    const float* w    = (const float*)d_in[2];
    float* out = (float*)d_out;

    const int Nnew = in_sizes[0] / KDIM;  // 8192
    const int Ntr  = in_sizes[1] / KDIM;  // 32768

    // Workspace layout (~20.2 MB): Xtr hi/lo f16, Xnew hi/lo f16, wy, exn.
    char* ws = (char*)d_ws;
    const size_t trsz = (size_t)Ntr * KDIM * sizeof(_Float16);   // 8 MB
    const size_t nwsz = (size_t)Nnew * KDIM * sizeof(_Float16);  // 2 MB
    _Float16* trh = (_Float16*)(ws);
    _Float16* trl = (_Float16*)(ws + trsz);
    _Float16* nwh = (_Float16*)(ws + 2 * trsz);
    _Float16* nwl = (_Float16*)(ws + 2 * trsz + nwsz);
    float*    wy  = (float*)   (ws + 2 * trsz + 2 * nwsz);
    float*    exn = (float*)   (ws + 2 * trsz + 2 * nwsz + (size_t)Ntr * sizeof(float));

    // Prep: one wave per row (32 threads x 4 f32 each = 128-dim row).
    prep_split_kernel<<<dim3((Ntr * 32 + 255) / 256), dim3(256), 0, stream>>>(
        Xtr, w, trh, trl, wy, Ntr);
    prep_split_kernel<<<dim3((Nnew * 32 + 255) / 256), dim3(256), 0, stream>>>(
        Xnew, nullptr, nwh, nwl, exn, Nnew);

    rbf_main_kernel<<<dim3(Nnew / 16), dim3(128), 0, stream>>>(
        nwh, nwl, trh, trl, exn, wy, out, Ntr);
}
